// PershomBase_51531017617852
// MI455X (gfx1250) — compile-verified, matrix operand order checked by reading, not run on hardware
//
#include <hip/hip_runtime.h>
#include <math.h>

typedef __attribute__((ext_vector_type(16))) _Float16 v16h;
typedef __attribute__((ext_vector_type(8)))  float    v8f;

#define BGRAPH 256
#define NPER   256
#define EPER   1024
#define DIN    256
#define HID    512
#define KC     64
#define CLS    10

// ---------------------------------------------------------------------------
// Kernel 1: filt = sigmoid(relu(X @ W1 + b1) @ W2 + b2), fully fused.
// 512 blocks x 256 threads (8 waves). 128 rows per block, 16 rows per wave.
// A-tile (128x256 f16) staged once; W1 N-tile (256x16 f16, transposed) staged
// per iteration and shared by all 8 waves. GEMM via v_wmma_f32_16x16x32_f16.
// ---------------------------------------------------------------------------
__global__ __launch_bounds__(256) void mlp_filt_kernel(
    const float* __restrict__ x, const float* __restrict__ W1,
    const float* __restrict__ b1, const float* __restrict__ W2,
    const float* __restrict__ b2, float* __restrict__ filt)
{
  __shared__ __align__(32) _Float16 Ah[128 * DIN];  // 64KB: A tile, [row][k]
  __shared__ __align__(32) _Float16 Bh[16 * DIN];   // 8KB: B tile, [n][k] (transposed)

  const int tid  = threadIdx.x;
  const int lane = tid & 31;
  const int wave = tid >> 5;
  const int rowBase = blockIdx.x * 128;

  // Stage A: fp32 -> fp16 into LDS (8192 float4 loads, coalesced)
  {
    const float4* xv = (const float4*)(x + (size_t)rowBase * DIN);
    #pragma unroll 4
    for (int p = 0; p < 32; ++p) {
      int f4 = p * 256 + tid;
      float4 vv = xv[f4];
      int base = f4 * 4;
      Ah[base + 0] = (_Float16)vv.x;
      Ah[base + 1] = (_Float16)vv.y;
      Ah[base + 2] = (_Float16)vv.z;
      Ah[base + 3] = (_Float16)vv.w;
    }
  }

  float racc[8];
  #pragma unroll
  for (int i = 0; i < 8; ++i) racc[i] = 0.f;

  const int waveRow = wave * 16;
  const int aRow    = waveRow + (lane & 15);
  const int kHalf   = (lane >> 4) << 4;    // lanes 16-31 hold K=16..31 of the chunk

  for (int n0 = 0; n0 < HID; n0 += 16) {
    __syncthreads();  // previous iteration done reading Bh
    // Stage W1[:, n0:n0+16] -> Bh[n][k] (f16)
    #pragma unroll
    for (int p = 0; p < 4; ++p) {
      int f4 = p * 256 + tid;              // 0..1023
      int k  = f4 >> 2;
      int jj = (f4 & 3) * 4;
      float4 vv = *(const float4*)(W1 + (size_t)k * HID + n0 + jj);
      Bh[(jj + 0) * DIN + k] = (_Float16)vv.x;
      Bh[(jj + 1) * DIN + k] = (_Float16)vv.y;
      Bh[(jj + 2) * DIN + k] = (_Float16)vv.z;
      Bh[(jj + 3) * DIN + k] = (_Float16)vv.w;
    }
    __syncthreads();

    v8f cacc = {};
    #pragma unroll
    for (int k0 = 0; k0 < DIN; k0 += 32) {
      v16h afrag = *(const v16h*)(&Ah[aRow * DIN + k0 + kHalf]);
      v16h bfrag = *(const v16h*)(&Bh[(lane & 15) * DIN + k0 + kHalf]);
      cacc = __builtin_amdgcn_wmma_f32_16x16x32_f16(
          false, afrag, false, bfrag, (short)0, cacc, false, false);
    }

    // fuse bias + relu + (.@W2) column reduction on the accumulator
    int n = n0 + (lane & 15);
    float b1v = b1[n];
    float w2v = W2[n];
    #pragma unroll
    for (int i = 0; i < 8; ++i) {
      float hv = fmaxf(cacc[i] + b1v, 0.f);
      racc[i] += hv * w2v;
    }
  }

  // reduce across the 16 lanes of each half-wave (columns of the tile)
  #pragma unroll
  for (int m = 1; m < 16; m <<= 1) {
    #pragma unroll
    for (int i = 0; i < 8; ++i) racc[i] += __shfl_xor(racc[i], m, 32);
  }

  float b2v = b2[0];
  int halfBase = (lane >> 4) * 8;  // lanes 0-15 -> rows 0..7; lanes 16-31 -> rows 8..15
  if ((lane & 15) < 8) {
    int i = lane & 15;
    float z = racc[i] + b2v;
    float s = 1.f / (1.f + __expf(-z));
    filt[rowBase + waveRow + halfBase + i] = s;
  }
}

// ---------------------------------------------------------------------------
// Kernel 2: per-graph persistence (bitonic sort + union-find scan) fused with
// rational-hat readout and the final 192x10 classifier. One block per graph.
// ---------------------------------------------------------------------------
__global__ __launch_bounds__(256) void persistence_kernel(
    const float* __restrict__ filt, const int* __restrict__ edges,
    const float* __restrict__ c0, const float* __restrict__ c1,
    const float* __restrict__ c2, const float* __restrict__ rr,
    const float* __restrict__ Wc, const float* __restrict__ bc,
    float* __restrict__ out)
{
  __shared__ unsigned long long keys[EPER];  // (ev_bits << 32) | edge_idx
  __shared__ unsigned int  uvArr[EPER];      // packed (v<<16)|u after sort
  __shared__ float         bArr[EPER];       // H0 birth per sorted edge
  __shared__ unsigned char flagArr[EPER];    // 1 = "same" (essential H1 edge)
  __shared__ float fLDS[NPER];
  __shared__ int   pLDS[NPER];
  __shared__ float partial[256];
  __shared__ float feat[3 * KC];

  const int tid = threadIdx.x;
  const int g   = blockIdx.x;
  const int* eg = edges + (size_t)g * EPER * 2;

  fLDS[tid] = filt[(size_t)g * NPER + tid];
  pLDS[tid] = tid;
  __syncthreads();

  // build sort keys: ev = max(f[u], f[v]); sigmoid > 0 so float order == uint order
  for (int e = tid; e < EPER; e += 256) {
    int u = eg[e * 2 + 0], v = eg[e * 2 + 1];
    float ev = fmaxf(fLDS[u], fLDS[v]);
    keys[e] = (((unsigned long long)__float_as_uint(ev)) << 32) | (unsigned)e;
  }

  // bitonic sort, ascending, stable via embedded index
  for (unsigned size = 2; size <= EPER; size <<= 1) {
    for (unsigned stride = size >> 1; stride >= 1; stride >>= 1) {
      __syncthreads();
      for (unsigned p = tid; p < EPER / 2; p += 256) {
        unsigned i = 2 * p - (p & (stride - 1));
        unsigned j = i + stride;
        bool asc = ((i & size) == 0);
        unsigned long long a = keys[i], b = keys[j];
        if ((a > b) == asc) { keys[i] = b; keys[j] = a; }
      }
    }
  }
  __syncthreads();

  for (int e = tid; e < EPER; e += 256) {
    unsigned idx = (unsigned)(keys[e] & 0xffffffffu);
    unsigned u = (unsigned)eg[idx * 2 + 0];
    unsigned v = (unsigned)eg[idx * 2 + 1];
    uvArr[e] = u | (v << 16);
  }
  __syncthreads();

  // serial union-find scan; parent kept fully compressed (vector relabel)
  for (int t = 0; t < EPER; ++t) {
    unsigned uv = uvArr[t];
    int u = uv & 0xffff, v = (int)(uv >> 16);
    int ru = pLDS[u], rv = pLDS[v];
    bool same = (ru == rv);
    float fu = fLDS[ru], fv = fLDS[rv];
    bool u_elder = (fu < fv) || ((fu == fv) && (ru <= rv));
    int elder   = u_elder ? ru : rv;
    int younger = u_elder ? rv : ru;
    float w = __uint_as_float((unsigned)(keys[t] >> 32));
    float birth = same ? w : fLDS[younger];
    __syncthreads();                         // reads of parent[] complete
    if (!same && pLDS[tid] == younger) pLDS[tid] = elder;
    if (tid == 0) {
      bArr[t]    = birth;
      flagArr[t] = same ? 1 : 0;
    }
    __syncthreads();                         // writes visible for next step
  }

  float rabs = fabsf(rr[0]);

  // f0: finite H0 pairs (birth, death) over merge edges
  {
    int k = tid >> 2, j = tid & 3;
    float ck0 = c0[k * 2 + 0], ck1 = c0[k * 2 + 1];
    float acc = 0.f;
    for (int e = j; e < EPER; e += 4) {
      if (!flagArr[e]) {
        float dd   = __uint_as_float((unsigned)(keys[e] >> 32));
        float dist = fabsf(bArr[e] - ck0) + fabsf(dd - ck1);
        acc += 1.f / (1.f + dist) - 1.f / (1.f + fabsf(rabs - dist));
      }
    }
    partial[tid] = acc;
    __syncthreads();
    if (tid < KC)
      feat[tid] = partial[4 * tid] + partial[4 * tid + 1] +
                  partial[4 * tid + 2] + partial[4 * tid + 3];
    __syncthreads();
  }
  // f0e: essential H0 (roots), points = filtration values
  {
    int k = tid >> 2, j = tid & 3;
    float ck = c1[k];
    float acc = 0.f;
    for (int i = j; i < NPER; i += 4) {
      if (pLDS[i] == i) {
        float dist = fabsf(fLDS[i] - ck);
        acc += 1.f / (1.f + dist) - 1.f / (1.f + fabsf(rabs - dist));
      }
    }
    partial[tid] = acc;
    __syncthreads();
    if (tid < KC)
      feat[KC + tid] = partial[4 * tid] + partial[4 * tid + 1] +
                       partial[4 * tid + 2] + partial[4 * tid + 3];
    __syncthreads();
  }
  // f1e: essential H1 (cycle-creating edges), points = edge values
  {
    int k = tid >> 2, j = tid & 3;
    float ck = c2[k];
    float acc = 0.f;
    for (int e = j; e < EPER; e += 4) {
      if (flagArr[e]) {
        float dist = fabsf(__uint_as_float((unsigned)(keys[e] >> 32)) - ck);
        acc += 1.f / (1.f + dist) - 1.f / (1.f + fabsf(rabs - dist));
      }
    }
    partial[tid] = acc;
    __syncthreads();
    if (tid < KC)
      feat[2 * KC + tid] = partial[4 * tid] + partial[4 * tid + 1] +
                           partial[4 * tid + 2] + partial[4 * tid + 3];
    __syncthreads();
  }

  // classifier: out[g] = feat @ Wc + bc
  if (tid < CLS) {
    float s = bc[tid];
    #pragma unroll 4
    for (int jn = 0; jn < 3 * KC; ++jn) s += feat[jn] * Wc[jn * CLS + tid];
    out[(size_t)g * CLS + tid] = s;
  }
}

// ---------------------------------------------------------------------------
extern "C" void kernel_launch(void* const* d_in, const int* in_sizes, int n_in,
                              void* d_out, int out_size, void* d_ws, size_t ws_size,
                              hipStream_t stream) {
  const float* x     = (const float*)d_in[0];
  const int*   edges = (const int*)d_in[1];
  const float* W1    = (const float*)d_in[2];
  const float* b1    = (const float*)d_in[3];
  const float* W2    = (const float*)d_in[4];
  const float* b2    = (const float*)d_in[5];
  const float* c0    = (const float*)d_in[6];
  const float* c1    = (const float*)d_in[7];
  const float* c2    = (const float*)d_in[8];
  const float* r     = (const float*)d_in[9];
  const float* Wc    = (const float*)d_in[10];
  const float* bc    = (const float*)d_in[11];

  float* filt = (float*)d_ws;  // BGRAPH*NPER = 65536 floats of scratch

  mlp_filt_kernel<<<(BGRAPH * NPER) / 128, 256, 0, stream>>>(x, W1, b1, W2, b2, filt);
  persistence_kernel<<<BGRAPH, 256, 0, stream>>>(filt, edges, c0, c1, c2, r, Wc, bc,
                                                 (float*)d_out);
}
